// GAT4_73778948211132
// MI455X (gfx1250) — matrix operand order repaired and checked by reading, not compiled, not run on hardware
//
#include <hip/hip_runtime.h>
#include <hip/hip_bf16.h>

typedef __bf16 bf16x16 __attribute__((ext_vector_type(16)));
typedef float  f32x8   __attribute__((ext_vector_type(8)));

#define BN_EPS 1e-5f
#define NB 8   // batch

// ---------------------------------------------------------------------------
// 3x3 pad-1 conv as implicit GEMM on v_wmma_f32_16x16x32_bf16 (wave32).
// K ordering is tap-major: kk = tap*Cin + cin (Cin power of two -> shift/mask
// decode). One wave computes NMT (1 or 2) 16-cout tiles x 16 pixels, sharing
// one B fragment. B loads are branch-free: coordinates are clamped (always
// legal) and the value is zeroed by cndmask -> loads issue as a clause.
// Both A fragments are loaded before the back-to-back WMMAs (no WAR nops).
// GUARD=true only for Cout==8 (half-wave-uniform bound check in epilogue).
// ---------------------------------------------------------------------------
template <int NMT, bool GUARD>
__global__ __launch_bounds__(256)
void conv3x3_wmma(const float* __restrict__ in, const float* __restrict__ wgt,
                  const float* __restrict__ bias, float* __restrict__ out,
                  int Cin, int Cout, int H, int W, int Kpad,
                  int cinLog, int wLog, int hwLog)
{
  __shared__ __bf16 smemA[NMT * 16 * 576];     // up to 36,864 bytes
  const int tid = threadIdx.x;
  const int m0  = blockIdx.y * (16 * NMT);
  const int cmask = Cin - 1;

  // cooperative weight-tile load, re-permuted to tap-major K (fp32 -> bf16)
  for (int idx = tid; idx < NMT * 16 * Kpad; idx += 256) {
    int m = idx / Kpad, k = idx - m * Kpad;
    int tap = k >> cinLog, cin = k & cmask;
    float v = 0.f;
    int mg = m0 + m;
    if (mg < Cout && tap < 9)
      v = wgt[(size_t)(mg * Cin + cin) * 9 + tap];
    smemA[m * Kpad + k] = (__bf16)v;
  }
  __syncthreads();

  const int wave = tid >> 5, lane = tid & 31;
  const int ntxLog = wLog - 4;
  const int nsp  = H * (W >> 4);
  const int tile = blockIdx.x * 8 + wave;
  if (tile >= nsp) return;                     // uniform per wave
  const int b  = blockIdx.z;
  const int y  = tile >> ntxLog;
  const int x0 = (tile & ((1 << ntxLog) - 1)) << 4;
  const int hh = lane >> 4;                    // lane half
  const int mn = lane & 15;                    // A row / B column
  const int x  = x0 + mn;
  const float* inB = in + ((size_t)b << (cinLog + hwLog));

  f32x8 acc[NMT];
#pragma unroll
  for (int mt = 0; mt < NMT; ++mt)
    acc[mt] = (f32x8){0.f,0.f,0.f,0.f,0.f,0.f,0.f,0.f};

  for (int k0 = 0; k0 < Kpad; k0 += 32) {
    // ---- B fragment: 16 unconditional (clamped) loads, longest latency first
    float fv[16];
#pragma unroll
    for (int i = 0; i < 16; ++i) {
      int kk  = k0 + 16 * hh + i;
      int tap = kk >> cinLog;
      int cin = kk & cmask;
      int tc  = tap > 8 ? 8 : tap;
      int dy  = (tc >= 3) + (tc >= 6);
      int dx  = tc - 3 * dy;
      int iy  = y + dy - 1;
      int ix  = x + dx - 1;
      bool ok = (tap <= 8) & ((unsigned)iy < (unsigned)H) & ((unsigned)ix < (unsigned)W);
      int iyc = iy < 0 ? 0 : (iy >= H ? H - 1 : iy);
      int ixc = ix < 0 ? 0 : (ix >= W ? W - 1 : ix);
      float v = inB[(cin << hwLog) + (iyc << wLog) + ixc];
      fv[i] = ok ? v : 0.f;
    }
    // ---- all A fragments up front (distinct regs -> no WMMA WAR hazard)
    bf16x16 afrag[NMT];
#pragma unroll
    for (int mt = 0; mt < NMT; ++mt) {
#pragma unroll
      for (int i = 0; i < 16; ++i) {
        int koff = 16 * hh + i;
        afrag[mt][i] = smemA[(mt * 16 + mn) * Kpad + k0 + koff];
      }
    }
    bf16x16 bfrag;
#pragma unroll
    for (int i = 0; i < 16; ++i) bfrag[i] = (__bf16)fv[i];

#pragma unroll
    for (int mt = 0; mt < NMT; ++mt)
      acc[mt] = __builtin_amdgcn_wmma_f32_16x16x32_bf16(false, afrag[mt], false, bfrag,
                                                        (short)0, acc[mt], false, false);
  }

  // ---- epilogue: vectorized bias fetch, straight-line stores when !GUARD
#pragma unroll
  for (int mt = 0; mt < NMT; ++mt) {
    int base = m0 + mt * 16 + 8 * hh;          // always a multiple of 8
    if (!GUARD || base < Cout) {
      const float4* b4 = (const float4*)(bias + base);
      float4 bv0 = b4[0], bv1 = b4[1];
      float ba[8] = {bv0.x, bv0.y, bv0.z, bv0.w, bv1.x, bv1.y, bv1.z, bv1.w};
      float* op = out + (((size_t)b * Cout + base) << hwLog) + (y << wLog) + x;
#pragma unroll
      for (int r = 0; r < 8; ++r)
        op[(size_t)r << hwLog] = acc[mt][r] + ba[r];
    }
  }
}

// ---------------------------------------------------------------------------
// training-mode BN: per-channel sum / sumsq reduction (plane is power of two)
// ---------------------------------------------------------------------------
__global__ __launch_bounds__(256)
void bn_stats_k(const float* __restrict__ x, float* __restrict__ stats,
                int C, int plane, int planeLog)
{
  const int c = blockIdx.y;
  __shared__ float ssum[256], ssq[256];
  float s = 0.f, q = 0.f;
  const int total = NB * plane;
  for (int i = blockIdx.x * 256 + threadIdx.x; i < total; i += gridDim.x * 256) {
    int b = i >> planeLog, pos = i & (plane - 1);
    float v = x[((size_t)b * C + c) * plane + pos];
    s += v; q += v * v;
  }
  ssum[threadIdx.x] = s; ssq[threadIdx.x] = q;
  __syncthreads();
  for (int st = 128; st > 0; st >>= 1) {
    if (threadIdx.x < st) {
      ssum[threadIdx.x] += ssum[threadIdx.x + st];
      ssq[threadIdx.x]  += ssq[threadIdx.x + st];
    }
    __syncthreads();
  }
  if (threadIdx.x == 0) {
    atomicAdd(&stats[c], ssum[0]);
    atomicAdd(&stats[C + c], ssq[0]);
  }
}

__global__ void bn_finalize_k(const float* __restrict__ stats,
                              const float* __restrict__ g,
                              const float* __restrict__ beta,
                              float* __restrict__ prm, int C, float inv_count)
{
  int c = threadIdx.x;
  if (c < C) {
    float mean = stats[c] * inv_count;
    float var  = stats[C + c] * inv_count - mean * mean;
    float sc   = g[c] * rsqrtf(var + BN_EPS);
    prm[c]     = sc;
    prm[C + c] = beta[c] - mean * sc;
  }
}

// relu(maxpool2(bn(x))) -- affine per tap (scale may be negative). Shapes pow2.
__global__ __launch_bounds__(256)
void bn_pool_relu_k(const float* __restrict__ x, const float* __restrict__ prm,
                    float* __restrict__ out, int C, int Hin, int Win,
                    int woLog, int hoLog, int cLog)
{
  const int Ho = Hin >> 1, Wo = Win >> 1;
  const int total = NB * C * Ho * Wo;
  for (int i = blockIdx.x * 256 + threadIdx.x; i < total; i += gridDim.x * 256) {
    int wo = i & (Wo - 1);
    int ho = (i >> woLog) & (Ho - 1);
    int c  = (i >> (woLog + hoLog)) & (C - 1);
    int b  = i >> (woLog + hoLog + cLog);
    const float* p = x + (((size_t)b * C + c) * Hin + 2 * ho) * Win + 2 * wo;
    float sc = prm[c], sh = prm[C + c];
    float a0 = p[0] * sc + sh,   a1 = p[1] * sc + sh;
    float a2 = p[Win] * sc + sh, a3 = p[Win + 1] * sc + sh;
    out[i] = fmaxf(fmaxf(fmaxf(a0, a1), fmaxf(a2, a3)), 0.f);
  }
}

// relu(upsample2_bilinear(bn(x))) [+ skip]; affine commutes with interp
__global__ __launch_bounds__(256)
void bn_up_relu_k(const float* __restrict__ x, const float* __restrict__ prm,
                  const float* __restrict__ skip, float* __restrict__ out,
                  int C, int Hin, int Win, int woLog, int hoLog, int cLog)
{
  const int Ho = Hin * 2, Wo = Win * 2;
  const int total = NB * C * Ho * Wo;
  for (int i = blockIdx.x * 256 + threadIdx.x; i < total; i += gridDim.x * 256) {
    int ox = i & (Wo - 1);
    int oy = (i >> woLog) & (Ho - 1);
    int c  = (i >> (woLog + hoLog)) & (C - 1);
    int b  = i >> (woLog + hoLog + cLog);
    float fy = (oy + 0.5f) * 0.5f - 0.5f;
    float fx = (ox + 0.5f) * 0.5f - 0.5f;
    int y0 = (int)floorf(fy); float wy = fy - y0;
    int x0 = (int)floorf(fx); float wx = fx - x0;
    int y0c = max(y0, 0), y1c = min(y0 + 1, Hin - 1);
    int x0c = max(x0, 0), x1c = min(x0 + 1, Win - 1);
    const float* p = x + ((size_t)b * C + c) * Hin * Win;
    float v00 = p[y0c * Win + x0c], v01 = p[y0c * Win + x1c];
    float v10 = p[y1c * Win + x0c], v11 = p[y1c * Win + x1c];
    float v = v00 * (1.f - wy) * (1.f - wx) + v01 * (1.f - wy) * wx
            + v10 * wy * (1.f - wx)        + v11 * wy * wx;
    v = fmaxf(v * prm[c] + prm[C + c], 0.f);
    if (skip) v += skip[i];
    out[i] = v;
  }
}

// relu(bn(x)) elementwise (GAT tail); plane power of two
__global__ __launch_bounds__(256)
void bn_relu_k(const float* __restrict__ x, const float* __restrict__ prm,
               float* __restrict__ out, int C, int plane, int planeLog)
{
  const int total = NB * C * plane;
  for (int i = blockIdx.x * 256 + threadIdx.x; i < total; i += gridDim.x * 256) {
    int c = (i >> planeLog) & (C - 1);
    out[i] = fmaxf(x[i] * prm[c] + prm[C + c], 0.f);
  }
}

// ---------------------------------------------------------------------------
// GAT block on the 16x16 grid (4-neighbour + self). One block per batch,
// one thread per node; proj / attention scores staged in LDS. Incoming edge
// sets recomputed from grid symmetry -> no atomics. Writes pre-BN output.
// ---------------------------------------------------------------------------
__global__ __launch_bounds__(256)
void gat_k(const float* __restrict__ x,          // [B,8,16,16]
           const float* __restrict__ reduce_w,   // [4,8]
           const float* __restrict__ lin_w,      // [16,4]
           const float* __restrict__ att_src,    // [4,4]
           const float* __restrict__ att_dst,    // [4,4]
           const float* __restrict__ gbias,      // [4]
           const float* __restrict__ restore_w,  // [8,4]
           float* __restrict__ out)              // [B,8,16,16] pre-BN
{
  const int b = blockIdx.x;
  const int n = threadIdx.x;          // node id 0..255
  const int ii = n >> 4, jj = n & 15;
  __shared__ float proj[256][16];
  __shared__ float as_[256][4], ad_[256][4];

  const float* xb = x + (size_t)b * 8 * 256;
  float xin[8];
#pragma unroll
  for (int c = 0; c < 8; ++c) xin[c] = xb[c * 256 + n];

  float xr[4];
#pragma unroll
  for (int cp = 0; cp < 4; ++cp) {
    float s = 0.f;
#pragma unroll
    for (int c = 0; c < 8; ++c) s += reduce_w[cp * 8 + c] * xin[c];
    xr[cp] = s;
  }
#pragma unroll
  for (int o = 0; o < 16; ++o) {
    float s = 0.f;
#pragma unroll
    for (int cp = 0; cp < 4; ++cp) s += lin_w[o * 4 + cp] * xr[cp];
    proj[n][o] = s;
  }
#pragma unroll
  for (int h = 0; h < 4; ++h) {
    float s = 0.f, d = 0.f;
#pragma unroll
    for (int c = 0; c < 4; ++c) {
      s += proj[n][h * 4 + c] * att_src[h * 4 + c];
      d += proj[n][h * 4 + c] * att_dst[h * 4 + c];
    }
    as_[n][h] = s; ad_[n][h] = d;
  }
  __syncthreads();

  int srcs[5]; int ns = 0;
  srcs[ns++] = n;
  if (ii > 0)  srcs[ns++] = n - 16;
  if (ii < 15) srcs[ns++] = n + 16;
  if (jj > 0)  srcs[ns++] = n - 1;
  if (jj < 15) srcs[ns++] = n + 1;

  float outn[4] = {0.f, 0.f, 0.f, 0.f};
  for (int h = 0; h < 4; ++h) {
    float e[5], mx = -1e30f;
    for (int k = 0; k < ns; ++k) {
      float v = as_[srcs[k]][h] + ad_[n][h];
      v = v > 0.f ? v : 0.2f * v;              // leaky_relu 0.2
      e[k] = v; mx = fmaxf(mx, v);
    }
    float den = 0.f;
    for (int k = 0; k < ns; ++k) { e[k] = __expf(e[k] - mx); den += e[k]; }
    float inv = 1.f / den;
#pragma unroll
    for (int c = 0; c < 4; ++c) {
      float a = 0.f;
      for (int k = 0; k < ns; ++k) a += e[k] * proj[srcs[k]][h * 4 + c];
      outn[c] += a * inv;
    }
  }
#pragma unroll
  for (int c = 0; c < 4; ++c) outn[c] = outn[c] * 0.25f + gbias[c];  // mean heads + bias

  float* ob = out + (size_t)b * 8 * 256;
#pragma unroll
  for (int c8 = 0; c8 < 8; ++c8) {
    float s = 0.f;
#pragma unroll
    for (int c = 0; c < 4; ++c) s += restore_w[c8 * 4 + c] * outn[c];
    ob[c8 * 256 + n] = s + xin[c8];            // restore conv + residual
  }
}

// ---------------------------------------------------------------------------
// host orchestration
// ---------------------------------------------------------------------------
static constexpr size_t OFF_CONV   = 0;                         // 67,108,864
static constexpr size_t OFF_T1     = 67108864;                  // 16,777,216
static constexpr size_t OFF_T2     = OFF_T1 + 16777216;         //  2,097,152
static constexpr size_t OFF_T3     = OFF_T2 + 2097152;          //    262,144
static constexpr size_t OFF_ACTA   = OFF_T3 + 262144;           //     65,536
static constexpr size_t OFF_GATPRE = OFF_ACTA + 65536;
static constexpr size_t OFF_ACTB   = OFF_GATPRE + 65536;
static constexpr size_t OFF_U4     = OFF_ACTB + 65536;          //    262,144
static constexpr size_t OFF_U3     = OFF_U4 + 262144;           //  2,097,152
static constexpr size_t OFF_U2     = OFF_U3 + 2097152;          // 16,777,216
static constexpr size_t OFF_STATS  = OFF_U2 + 16777216;         //        512
static constexpr size_t OFF_PRM    = OFF_STATS + 512;

static inline int ilog2(int v) { int r = 0; while ((1 << r) < v) ++r; return r; }

extern "C" void kernel_launch(void* const* d_in, const int* in_sizes, int n_in,
                              void* d_out, int out_size, void* d_ws, size_t ws_size,
                              hipStream_t stream)
{
  (void)in_sizes; (void)n_in; (void)out_size; (void)ws_size;
  const float* x = (const float*)d_in[0];
  char* ws = (char*)d_ws;
  float* convbuf = (float*)(ws + OFF_CONV);
  float* t1      = (float*)(ws + OFF_T1);
  float* t2      = (float*)(ws + OFF_T2);
  float* t3      = (float*)(ws + OFF_T3);
  float* actA    = (float*)(ws + OFF_ACTA);
  float* gatPre  = (float*)(ws + OFF_GATPRE);
  float* actB    = (float*)(ws + OFF_ACTB);
  float* sU4     = (float*)(ws + OFF_U4);
  float* sU3     = (float*)(ws + OFF_U3);
  float* sU2     = (float*)(ws + OFF_U2);
  float* stats   = (float*)(ws + OFF_STATS);
  float* prm     = (float*)(ws + OFF_PRM);

  auto conv = [&](const float* src, int Cin, int Cout, int H, int W, int wi) {
    int Kpad = (Cin * 9 + 31) & ~31;
    int nsp = H * (W >> 4);
    int cinLog = ilog2(Cin), wLog = ilog2(W), hwLog = ilog2(H) + wLog;
    const float* wp = (const float*)d_in[wi];
    const float* bp = (const float*)d_in[wi + 1];
    if (Cout >= 32) {
      dim3 g((nsp + 7) / 8, Cout / 32, NB);
      conv3x3_wmma<2, false><<<g, 256, 0, stream>>>(src, wp, bp, convbuf, Cin, Cout,
                                                    H, W, Kpad, cinLog, wLog, hwLog);
    } else if (Cout == 16) {
      dim3 g((nsp + 7) / 8, 1, NB);
      conv3x3_wmma<1, false><<<g, 256, 0, stream>>>(src, wp, bp, convbuf, Cin, Cout,
                                                    H, W, Kpad, cinLog, wLog, hwLog);
    } else {  // Cout == 8
      dim3 g((nsp + 7) / 8, 1, NB);
      conv3x3_wmma<1, true><<<g, 256, 0, stream>>>(src, wp, bp, convbuf, Cin, Cout,
                                                   H, W, Kpad, cinLog, wLog, hwLog);
    }
  };
  auto bn = [&](const float* src, int C, int plane, int gi) {
    hipMemsetAsync(stats, 0, 2 * C * sizeof(float), stream);
    int total = NB * plane;
    int nb = (total + 255) / 256; if (nb > 64) nb = 64;
    bn_stats_k<<<dim3(nb, C), 256, 0, stream>>>(src, stats, C, plane, ilog2(plane));
    bn_finalize_k<<<1, 64, 0, stream>>>(stats, (const float*)d_in[gi],
                                        (const float*)d_in[gi + 1], prm, C,
                                        1.f / (float)total);
  };
  auto blocks = [](int total) { return (total + 255) / 256; };

  // ---------------- encoder ----------------
  conv(x, 64, 32, 256, 256, 1);  bn(convbuf, 32, 65536, 3);
  bn_pool_relu_k<<<blocks(NB * 32 * 128 * 128), 256, 0, stream>>>(
      convbuf, prm, t1, 32, 256, 256, 7, 7, 5);

  conv(t1, 32, 16, 128, 128, 5); bn(convbuf, 16, 16384, 7);
  bn_pool_relu_k<<<blocks(NB * 16 * 64 * 64), 256, 0, stream>>>(
      convbuf, prm, t2, 16, 128, 128, 6, 6, 4);

  conv(t2, 16, 8, 64, 64, 9);    bn(convbuf, 8, 4096, 11);
  bn_pool_relu_k<<<blocks(NB * 8 * 32 * 32), 256, 0, stream>>>(
      convbuf, prm, t3, 8, 64, 64, 5, 5, 3);

  conv(t3, 8, 8, 32, 32, 13);    bn(convbuf, 8, 1024, 15);
  bn_pool_relu_k<<<blocks(NB * 8 * 16 * 16), 256, 0, stream>>>(
      convbuf, prm, actA, 8, 32, 32, 4, 4, 3);

  // ---------------- GAT bottleneck ----------------
  gat_k<<<NB, 256, 0, stream>>>(actA,
                                (const float*)d_in[17], (const float*)d_in[18],
                                (const float*)d_in[19], (const float*)d_in[20],
                                (const float*)d_in[21], (const float*)d_in[22],
                                gatPre);
  bn(gatPre, 8, 256, 23);
  bn_relu_k<<<blocks(NB * 8 * 256), 256, 0, stream>>>(gatPre, prm, actB, 8, 256, 8);

  // ---------------- decoder ----------------
  conv(actB, 8, 8, 16, 16, 25);  bn(convbuf, 8, 256, 27);
  bn_up_relu_k<<<blocks(NB * 8 * 32 * 32), 256, 0, stream>>>(
      convbuf, prm, t3, sU4, 8, 16, 16, 5, 5, 3);

  conv(sU4, 8, 16, 32, 32, 29);  bn(convbuf, 16, 1024, 31);
  bn_up_relu_k<<<blocks(NB * 16 * 64 * 64), 256, 0, stream>>>(
      convbuf, prm, t2, sU3, 16, 32, 32, 6, 6, 4);

  conv(sU3, 16, 32, 64, 64, 33); bn(convbuf, 32, 4096, 35);
  bn_up_relu_k<<<blocks(NB * 32 * 128 * 128), 256, 0, stream>>>(
      convbuf, prm, t1, sU2, 32, 64, 64, 7, 7, 5);

  conv(sU2, 32, 64, 128, 128, 37); bn(convbuf, 64, 16384, 39);
  bn_up_relu_k<<<blocks(NB * 64 * 256 * 256), 256, 0, stream>>>(
      convbuf, prm, nullptr, (float*)d_out, 64, 128, 128, 8, 8, 6);
}
